// HGTRModel_85203561218772
// MI455X (gfx1250) — matrix operand order repaired and checked by reading, not compiled, not run on hardware
//
#include <hip/hip_runtime.h>
#include <hip/hip_bf16.h>
#include <math.h>

// ---------------- model constants ----------------
#define NN    20000
#define EE    160000
#define BB    16
#define FIN   64
#define PROJD 256
#define HID   128
#define NHD   4        // heads in HGT
#define DD    32       // head dim
#define NL    2
#define NMAXB (NN/BB)  // 1250 nodes per graph (batch is sorted & even)
#define JKC   (NL*HID) // 256

typedef __attribute__((ext_vector_type(16))) __bf16 v16bf;
typedef __attribute__((ext_vector_type(8)))  float  v8f;

__device__ __forceinline__ float gelu_exact(float x) {
    return 0.5f * x * (1.0f + erff(x * 0.70710678118654752440f));
}

// order-preserving float<->uint for atomicMax-based segment max
__device__ __forceinline__ unsigned fenc(float f) {
    unsigned u = __float_as_uint(f);
    return (u & 0x80000000u) ? ~u : (u | 0x80000000u);
}
__device__ __forceinline__ float fdec(unsigned u) {
    return __uint_as_float((u & 0x80000000u) ? (u & 0x7FFFFFFFu) : ~u);
}

// ---- WMMA fragment loaders (CDNA5 16-bit A layout; full tiles, no guards) ----
// A fragment: lane half 'hi' reads K = base..base+7 and base+16..base+23, base = hi*8.
__device__ __forceinline__ v16bf load_afrag(const float* __restrict__ p /* row ptr + k0 + abase */) {
    const float4 x0 = ((const float4*)p)[0];
    const float4 x1 = ((const float4*)p)[1];
    const float4 x2 = ((const float4*)(p + 16))[0];
    const float4 x3 = ((const float4*)(p + 16))[1];
    v16bf a;
    a[0]=(__bf16)x0.x;  a[1]=(__bf16)x0.y;  a[2]=(__bf16)x0.z;  a[3]=(__bf16)x0.w;
    a[4]=(__bf16)x1.x;  a[5]=(__bf16)x1.y;  a[6]=(__bf16)x1.z;  a[7]=(__bf16)x1.w;
    a[8]=(__bf16)x2.x;  a[9]=(__bf16)x2.y;  a[10]=(__bf16)x2.z; a[11]=(__bf16)x2.w;
    a[12]=(__bf16)x3.x; a[13]=(__bf16)x3.y; a[14]=(__bf16)x3.z; a[15]=(__bf16)x3.w;
    return a;
}
// B fragment: element i = W[k0 + hi*8 + i + (i>=8 ? 8 : 0)][col]; lanes of a half read consecutive cols.
__device__ __forceinline__ v16bf load_bfrag(const float* __restrict__ Wc, int ldw) {
    v16bf b;
    #pragma unroll
    for (int i = 0; i < 16; ++i) b[i] = (__bf16)Wc[(size_t)(i + (i >= 8 ? 8 : 0)) * ldw];
    return b;
}

// ---------------- generic WMMA GEMM ----------------
// C[M,Nc] = act(A[M,K](lda) @ W[K,Nc](ldw) + bias)
// One wave computes a 32x32 macro-tile = 2x2 WMMA tiles (4 v_wmma per k-step).
// REQUIRES: M%32==0, K%32==0, Nc%32==0 (true for every call in this model).
// act: 0 = none, 1 = exact GELU
__global__ void wmma_gemm(const float* __restrict__ A, int lda,
                          const float* __restrict__ W, int ldw,
                          const float* __restrict__ bias,
                          float* __restrict__ C, int ldc,
                          int K, int act)
{
    const int lane = threadIdx.x;       // 0..31 (wave32)
    const int m0 = blockIdx.x * 32;
    const int n0 = blockIdx.y * 32;
    const int hi  = lane >> 4;
    const int l15 = lane & 15;
    const int abase = hi * 8;

    v8f acc00 = {0.f,0.f,0.f,0.f,0.f,0.f,0.f,0.f};
    v8f acc01 = acc00, acc10 = acc00, acc11 = acc00;

    const float* Arow0 = A + (size_t)(m0 + l15) * lda + abase;
    const float* Arow1 = A + (size_t)(m0 + 16 + l15) * lda + abase;
    const float* Wc0   = W + n0 + l15;        // N-tile 0 column
    const float* Wc1   = W + n0 + 16 + l15;   // N-tile 1 column

    for (int k0 = 0; k0 < K; k0 += 32) {
        v16bf a0 = load_afrag(Arow0 + k0);
        v16bf a1 = load_afrag(Arow1 + k0);
        const float* Wk = W ? (Wc0 + (size_t)(k0 + hi * 8) * ldw) : Wc0;
        v16bf b0 = load_bfrag(Wk, ldw);
        v16bf b1 = load_bfrag(Wk + 16, ldw);
        acc00 = __builtin_amdgcn_wmma_f32_16x16x32_bf16(false, a0, false, b0, (short)0, acc00, false, false);
        acc01 = __builtin_amdgcn_wmma_f32_16x16x32_bf16(false, a0, false, b1, (short)0, acc01, false, false);
        acc10 = __builtin_amdgcn_wmma_f32_16x16x32_bf16(false, a1, false, b0, (short)0, acc10, false, false);
        acc11 = __builtin_amdgcn_wmma_f32_16x16x32_bf16(false, a1, false, b1, (short)0, acc11, false, false);
    }

    float bv0 = bias ? bias[n0 + l15]      : 0.f;
    float bv1 = bias ? bias[n0 + 16 + l15] : 0.f;
    #pragma unroll
    for (int r = 0; r < 8; ++r) {
        int rr0 = m0 + r + hi * 8;          // C/D layout: VGPR r -> M=r (+8 for upper lanes)
        int rr1 = rr0 + 16;
        float v00 = acc00[r] + bv0, v01 = acc01[r] + bv1;
        float v10 = acc10[r] + bv0, v11 = acc11[r] + bv1;
        if (act == 1) { v00 = gelu_exact(v00); v01 = gelu_exact(v01);
                        v10 = gelu_exact(v10); v11 = gelu_exact(v11); }
        C[(size_t)rr0 * ldc + n0 + l15]      = v00;
        C[(size_t)rr0 * ldc + n0 + 16 + l15] = v01;
        C[(size_t)rr1 * ldc + n0 + l15]      = v10;
        C[(size_t)rr1 * ldc + n0 + 16 + l15] = v11;
    }
}

// ---------------- utility fills ----------------
__global__ void fill_f32(float* p, float v, int n) {
    int i = blockIdx.x * blockDim.x + threadIdx.x;
    if (i < n) p[i] = v;
}
__global__ void fill_u32(unsigned* p, unsigned v, int n) {
    int i = blockIdx.x * blockDim.x + threadIdx.x;
    if (i < n) p[i] = v;
}
__global__ void gelu_ip(float* x, int n) {
    int i = blockIdx.x * blockDim.x + threadIdx.x;
    if (i < n) x[i] = gelu_exact(x[i]);
}

// ---------------- edge attention ----------------
// logits[e,h] = dot(q[dst][di], kr[si]) * p_rel[h] / sqrt(D);  atomic segment max
__global__ void edge_logits_max(const int* __restrict__ ei,
                                const float* __restrict__ qdst, int qld,
                                const float* __restrict__ krs,
                                const float* __restrict__ prel,
                                float* __restrict__ logits,
                                unsigned* __restrict__ mbuf)
{
    int idx = blockIdx.x * blockDim.x + threadIdx.x;
    if (idx >= EE * NHD) return;
    int e = idx >> 2, h = idx & 3;
    int si = ei[e], di = ei[EE + e];
    const float* qp = qdst + (size_t)di * qld + h * DD;
    const float* kp = krs  + (size_t)si * HID + h * DD;
    float s = 0.f;
    #pragma unroll
    for (int d = 0; d < DD; ++d) s += qp[d] * kp[d];
    float lg = s * prel[h] * 0.17677669529663688f;  // 1/sqrt(32)
    logits[idx] = lg;
    atomicMax(mbuf + (size_t)di * NHD + h, fenc(lg));
}

__global__ void edge_expsum(const int* __restrict__ ei,
                            float* __restrict__ logits,
                            const unsigned* __restrict__ mbuf,
                            float* __restrict__ sbuf)
{
    int idx = blockIdx.x * blockDim.x + threadIdx.x;
    if (idx >= EE * NHD) return;
    int e = idx >> 2, h = idx & 3;
    int di = ei[EE + e];
    float m  = fdec(mbuf[(size_t)di * NHD + h]);
    float ex = expf(logits[idx] - m);
    logits[idx] = ex;
    atomicAdd(sbuf + (size_t)di * NHD + h, ex);
}

__global__ void edge_scatter(const int* __restrict__ ei,
                             const float* __restrict__ logits,
                             const float* __restrict__ sbuf,
                             const float* __restrict__ vrs,
                             float* __restrict__ agg)
{
    int idx = blockIdx.x * blockDim.x + threadIdx.x;
    if (idx >= EE * NHD) return;
    int e = idx >> 2, h = idx & 3;
    int si = ei[e], di = ei[EE + e];
    float a = logits[idx] / sbuf[(size_t)di * NHD + h];
    const float* vp = vrs + (size_t)si * HID + h * DD;
    float* ap = agg + (size_t)di * HID + h * DD;
    #pragma unroll
    for (int d = 0; d < DD; ++d) atomicAdd(ap + d, a * vp[d]);
}

// ---------------- gated skip + LayerNorm -> JK buffer ----------------
__global__ void skip_ln(const float* __restrict__ o,
                        const float* __restrict__ xprev, int xprev_ld,
                        const float* __restrict__ skipp,   // null => no skip
                        const float* __restrict__ g, const float* __restrict__ bta,
                        float* __restrict__ xjk, int loff)
{
    int n = blockIdx.x, c = threadIdx.x;   // 128 threads
    float v = o[(size_t)n * HID + c];
    if (skipp) {
        float al = 1.f / (1.f + expf(-(*skipp)));
        v = al * v + (1.f - al) * xprev[(size_t)n * xprev_ld + c];
    }
    __shared__ float red[HID];
    red[c] = v; __syncthreads();
    for (int s = 64; s > 0; s >>= 1) { if (c < s) red[c] += red[c + s]; __syncthreads(); }
    float mean = red[0] * (1.f / HID); __syncthreads();
    float d = v - mean;
    red[c] = d * d; __syncthreads();
    for (int s = 64; s > 0; s >>= 1) { if (c < s) red[c] += red[c + s]; __syncthreads(); }
    float var = red[0] * (1.f / HID);
    xjk[(size_t)n * JKC + loff + c] = d * rsqrtf(var + 1e-5f) * g[c] + bta[c];
}

// ---------------- GMT pooling pieces ----------------
// y[c] = b[c] + sum_j x[j] * W[j,c]   (1x128 vector times 128x128)
__global__ void smallvec_lin(const float* __restrict__ x,
                             const float* __restrict__ W,
                             const float* __restrict__ b,
                             float* __restrict__ y)
{
    int c = threadIdx.x;
    float a = b[c];
    for (int j = 0; j < HID; ++j) a += x[j] * W[j * HID + c];
    y[c] = a;
}

// PMA1 attention: out[b,:] = Qp + softmax_k(Qp.Kp[b,k]/sqrt(128)) @ Vp[b]
__global__ void pma1_attn(const float* __restrict__ qp1,
                          const float* __restrict__ Kp,
                          const float* __restrict__ Vp,
                          float* __restrict__ outb)
{
    int b = blockIdx.x, t = threadIdx.x;   // 256 threads
    __shared__ float sl[NMAXB];
    __shared__ float red[256];
    float lmax = -1e30f;
    for (int k = t; k < NMAXB; k += 256) {
        const float* kp = Kp + ((size_t)b * NMAXB + k) * HID;
        float s = 0.f;
        for (int j = 0; j < HID; ++j) s += qp1[j] * kp[j];
        s *= 0.08838834764831845f;         // 1/sqrt(128)
        sl[k] = s; lmax = fmaxf(lmax, s);
    }
    red[t] = lmax; __syncthreads();
    for (int s = 128; s > 0; s >>= 1) { if (t < s) red[t] = fmaxf(red[t], red[t + s]); __syncthreads(); }
    float mx = red[0]; __syncthreads();
    float lsum = 0.f;
    for (int k = t; k < NMAXB; k += 256) { float ex = expf(sl[k] - mx); sl[k] = ex; lsum += ex; }
    red[t] = lsum; __syncthreads();
    for (int s = 128; s > 0; s >>= 1) { if (t < s) red[t] += red[t + s]; __syncthreads(); }
    float ssum = red[0]; __syncthreads();
    if (t < HID) {
        float o = 0.f;
        for (int k = 0; k < NMAXB; ++k) o += sl[k] * Vp[((size_t)b * NMAXB + k) * HID + t];
        outb[b * HID + t] = qp1[t] + o / ssum;
    }
}

// out[b, outoff+c] = in[b,c] + relu(bo[c] + sum_j in[b,j]*Wo[j,c])   (in-place safe)
__global__ void mab_fco(const float* __restrict__ inb,
                        const float* __restrict__ Wo, const float* __restrict__ bo,
                        float* __restrict__ outb, int outld, int outoff)
{
    int b = blockIdx.x, c = threadIdx.x;   // 128 threads
    __shared__ float hsh[HID];
    hsh[c] = inb[b * HID + c]; __syncthreads();
    float a = bo[c];
    for (int j = 0; j < HID; ++j) a += hsh[j] * Wo[j * HID + c];
    outb[(size_t)b * outld + outoff + c] = hsh[c] + fmaxf(a, 0.f);
}

// SAB with lk=1: softmax is identity => out = (h@Wq+bq) + (h@Wv+bv)
__global__ void sab_qv(const float* __restrict__ hin,
                       const float* __restrict__ Wq, const float* __restrict__ bq,
                       const float* __restrict__ Wv, const float* __restrict__ bv,
                       float* __restrict__ outb)
{
    int b = blockIdx.x, c = threadIdx.x;
    __shared__ float hsh[HID];
    hsh[c] = hin[b * HID + c]; __syncthreads();
    float a = bq[c] + bv[c];
    for (int j = 0; j < HID; ++j) a += hsh[j] * (Wq[j * HID + c] + Wv[j * HID + c]);
    outb[b * HID + c] = a;
}

// PMA2 with lk=1: out = qp2 + (h@Wv+bv)
__global__ void pma2_qv(const float* __restrict__ hin,
                        const float* __restrict__ Wv, const float* __restrict__ bv,
                        const float* __restrict__ qp2, float* __restrict__ outb)
{
    int b = blockIdx.x, c = threadIdx.x;
    __shared__ float hsh[HID];
    hsh[c] = hin[b * HID + c]; __syncthreads();
    float a = qp2[c] + bv[c];
    for (int j = 0; j < HID; ++j) a += hsh[j] * Wv[j * HID + c];
    outb[b * HID + c] = a;
}

// ---------------- y-MLP, gmlp ----------------
__global__ void ymlp_k(const float* __restrict__ yb,
                       const float* __restrict__ W1, const float* __restrict__ b1,
                       const float* __restrict__ W2, const float* __restrict__ b2,
                       float* __restrict__ hcat)
{
    __shared__ float y1[BB][16];
    int t = threadIdx.x;                 // 256 threads: b = t/16, j = t%16
    int b = t >> 4, j = t & 15;
    float v = fmaxf(b1[j] + yb[b] * W1[j], 0.f);
    y1[b][j] = v; __syncthreads();
    float a = b2[j];
    for (int i = 0; i < 16; ++i) a += y1[b][i] * W2[i * 16 + j];
    hcat[b * 272 + 256 + j] = fmaxf(a, 0.f);
}

// gelu(batch_norm(A@W+b)) with stats over the 16 rows; one thread per column
__global__ void gmlp_layer(const float* __restrict__ A, int Kc,
                           const float* __restrict__ W, const float* __restrict__ bias,
                           const float* __restrict__ gg, const float* __restrict__ bb,
                           float* __restrict__ C, int Nc)
{
    int c = blockIdx.x * blockDim.x + threadIdx.x;
    if (c >= Nc) return;
    float vals[BB];
    float mean = 0.f;
    for (int b = 0; b < BB; ++b) {
        float a = bias[c];
        for (int j = 0; j < Kc; ++j) a += A[b * Kc + j] * W[j * Nc + c];
        vals[b] = a; mean += a;
    }
    mean *= (1.f / BB);
    float var = 0.f;
    for (int b = 0; b < BB; ++b) { float d = vals[b] - mean; var += d * d; }
    var *= (1.f / BB);
    float rs = rsqrtf(var + 1e-5f);
    for (int b = 0; b < BB; ++b)
        C[b * Nc + c] = gelu_exact((vals[b] - mean) * rs * gg[c] + bb[c]);
}

__global__ void gmlp_final(const float* __restrict__ A,
                           const float* __restrict__ W4, const float* __restrict__ b4,
                           float* __restrict__ out)
{
    int b = threadIdx.x;
    if (b >= BB) return;
    float a = b4[0];
    for (int j = 0; j < 64; ++j) a += A[b * 64 + j] * W4[j];
    out[b] = a;
}

// ---------------- parameter leaf indexing (jax sorted-key tree order) ----------------
enum {
    IN_XOP = 0, IN_XVAR, IN_EIOV, IN_EIVO, IN_BOP, IN_BVAR, IN_YBASE,
    PB = 7,
    G_W1 = PB, G_W2, G_W3, G_W4, G_b1, G_b2, G_b3, G_b4, G_be1, G_be2, G_be3, G_g1, G_g2, G_g3,
    LAY0 = PB + 14,          // 20 leaves per layer
    NODELIN = PB + 54,       // op.W, op.b, var.W, var.b
    POOLB = PB + 58,         // 26 leaves per node type
    PROJB = PB + 110,        // op.W, op.b, var.W, var.b
    YMLPB = PB + 114         // W1, W2, b1, b2
};
enum { L_KREL_OV = 0, L_KREL_VO, L_KQV_OP_W, L_KQV_OP_B, L_KQV_VAR_W, L_KQV_VAR_B,
       L_LN_OP_B, L_LN_OP_G, L_LN_VAR_B, L_LN_VAR_G,
       L_OUT_OP_W, L_OUT_OP_B, L_OUT_VAR_W, L_OUT_VAR_B,
       L_PREL_OV, L_PREL_VO, L_SKIP_OP, L_SKIP_VAR, L_VREL_OV, L_VREL_VO };
enum { PL_S1 = 0, PL_S2,
       PL_P1_WK, PL_P1_WO, PL_P1_WQ, PL_P1_WV, PL_P1_BK, PL_P1_BO, PL_P1_BQ, PL_P1_BV,
       PL_P2_WK, PL_P2_WO, PL_P2_WQ, PL_P2_WV, PL_P2_BK, PL_P2_BO, PL_P2_BQ, PL_P2_BV,
       PL_SB_WK, PL_SB_WO, PL_SB_WQ, PL_SB_WV, PL_SB_BK, PL_SB_BO, PL_SB_BQ, PL_SB_BV };

extern "C" void kernel_launch(void* const* d_in, const int* in_sizes, int n_in,
                              void* d_out, int out_size, void* d_ws, size_t ws_size,
                              hipStream_t stream)
{
    auto f = [&](int i) { return (const float*)d_in[i]; };
    auto u = [&](int i) { return (const int*)d_in[i]; };

    // ---- workspace layout (floats); ~230 MB total, aliased where legal ----
    float* ws = (float*)d_ws;
    size_t off = 0;
    auto alloc = [&](size_t n) { float* p = ws + off; off += n; return p; };
    float* xproj[2] = { alloc((size_t)NN * PROJD), alloc((size_t)NN * PROJD) };
    float* xjk[2]   = { alloc((size_t)NN * JKC),   alloc((size_t)NN * JKC)   };
    float* kqvb[2]  = { alloc((size_t)NN * 384),   alloc((size_t)NN * 384)   };
    float* krb[2]   = { alloc((size_t)NN * HID),   alloc((size_t)NN * HID)   };
    float* vrb[2]   = { alloc((size_t)NN * HID),   alloc((size_t)NN * HID)   };
    float* logitsb  = alloc((size_t)EE * NHD);
    unsigned* mbuf  = (unsigned*)alloc((size_t)NN * NHD);
    float* sbuf     = alloc((size_t)NN * NHD);
    float* aggb[2]  = { alloc((size_t)NN * HID), alloc((size_t)NN * HID) };
    float* xfinb[2] = { alloc((size_t)NN * HID), alloc((size_t)NN * HID) };
    float* qp1  = alloc(HID);
    float* qp2  = alloc(HID);
    float* h1   = alloc(BB * HID);
    float* h2   = alloc(BB * HID);
    float* hcat = alloc(BB * 272);
    float* g1b  = alloc(BB * 256);
    float* g2b  = alloc(BB * 128);
    float* g3b  = alloc(BB * 64);
    // aliases: kqv buffers are dead after attention / after layers
    float* obuf[2] = { kqvb[0], kqvb[1] };                      // out-proj result
    float* Kp = kqvb[0];                                        // pool phase
    float* Vp = kqvb[0] + (size_t)NN * HID;
    (void)in_sizes; (void)n_in; (void)out_size; (void)ws_size;

    auto gemm = [&](const float* A, int lda, const float* W, int ldw,
                    const float* bias, float* C, int ldc,
                    int M, int K, int Nc, int act) {
        dim3 g(M / 32, Nc / 32);        // full 32x32 macro-tiles (all dims verified)
        wmma_gemm<<<g, dim3(32), 0, stream>>>(A, lda, W, ldw, bias, C, ldc, K, act);
    };

    // ---- stage A: per-type projection + GELU ----
    for (int t = 0; t < 2; ++t)
        gemm(f(t == 0 ? IN_XOP : IN_XVAR), FIN,
             f(PROJB + 2 * t), PROJD, f(PROJB + 2 * t + 1),
             xproj[t], PROJD, NN, FIN, PROJD, 1);

    const int totEH = EE * NHD;
    const int ebl = (totEH + 255) / 256;

    // ---- stage B: 2 HGT layers ----
    for (int l = 0; l < NL; ++l) {
        const int LB = LAY0 + l * 20;
        const int cin = (l == 0) ? PROJD : HID;
        // fused K/Q/V projection per node type
        for (int t = 0; t < 2; ++t) {
            const float* Ain = (l == 0) ? xproj[t] : xjk[t];   // layer1 reads JK cols 0..127
            int lda = (l == 0) ? PROJD : JKC;
            gemm(Ain, lda, f(LB + (t == 0 ? L_KQV_OP_W : L_KQV_VAR_W)), 384,
                 f(LB + (t == 0 ? L_KQV_OP_B : L_KQV_VAR_B)),
                 kqvb[t], 384, NN, cin, 384, 0);
        }
        // zero aggregation buffers
        for (int t = 0; t < 2; ++t)
            fill_f32<<<(NN * HID + 255) / 256, 256, 0, stream>>>(aggb[t], 0.f, NN * HID);
        // per-head relation transforms kr/vr (WMMA, K=32, Nc=32)
        for (int r = 0; r < 2; ++r) {
            int src = (r == 0) ? 0 : 1;
            const float* krel = f(LB + (r == 0 ? L_KREL_OV : L_KREL_VO));
            const float* vrel = f(LB + (r == 0 ? L_VREL_OV : L_VREL_VO));
            for (int h = 0; h < NHD; ++h) {
                gemm(kqvb[src] + h * DD, 384, krel + h * DD * DD, DD, nullptr,
                     krb[r] + h * DD, HID, NN, DD, DD, 0);
                gemm(kqvb[src] + 2 * HID + h * DD, 384, vrel + h * DD * DD, DD, nullptr,
                     vrb[r] + h * DD, HID, NN, DD, DD, 0);
            }
        }
        // edge attention per relation (rel0: op->var, rel1: var->op)
        for (int r = 0; r < 2; ++r) {
            int dst = (r == 0) ? 1 : 0;
            const int* ei = u(r == 0 ? IN_EIOV : IN_EIVO);
            const float* prel = f(LB + (r == 0 ? L_PREL_OV : L_PREL_VO));
            fill_u32<<<(NN * NHD + 255) / 256, 256, 0, stream>>>(mbuf, 0u, NN * NHD);
            fill_f32<<<(NN * NHD + 255) / 256, 256, 0, stream>>>(sbuf, 0.f, NN * NHD);
            edge_logits_max<<<ebl, 256, 0, stream>>>(ei, kqvb[dst] + HID, 384,
                                                     krb[r], prel, logitsb, mbuf);
            edge_expsum<<<ebl, 256, 0, stream>>>(ei, logitsb, mbuf, sbuf);
            edge_scatter<<<ebl, 256, 0, stream>>>(ei, logitsb, sbuf, vrb[r], aggb[dst]);
        }
        // out projection: gelu(agg) @ Wout + b, then gated skip + LN into JK buffer
        for (int t = 0; t < 2; ++t)
            gelu_ip<<<(NN * HID + 255) / 256, 256, 0, stream>>>(aggb[t], NN * HID);
        for (int t = 0; t < 2; ++t)
            gemm(aggb[t], HID, f(LB + (t == 0 ? L_OUT_OP_W : L_OUT_VAR_W)), HID,
                 f(LB + (t == 0 ? L_OUT_OP_B : L_OUT_VAR_B)),
                 obuf[t], HID, NN, HID, HID, 0);
        for (int t = 0; t < 2; ++t) {
            const float* skipp = (l == 0) ? nullptr
                                          : f(LB + (t == 0 ? L_SKIP_OP : L_SKIP_VAR));
            skip_ln<<<NN, HID, 0, stream>>>(obuf[t], xjk[t], JKC, skipp,
                                            f(LB + (t == 0 ? L_LN_OP_G : L_LN_VAR_G)),
                                            f(LB + (t == 0 ? L_LN_OP_B : L_LN_VAR_B)),
                                            xjk[t], l * HID);
        }
    }

    // ---- stage C: JumpingKnowledge cat -> node_lin + GELU ----
    for (int t = 0; t < 2; ++t)
        gemm(xjk[t], JKC, f(NODELIN + 2 * t), HID, f(NODELIN + 2 * t + 1),
             xfinb[t], HID, NN, JKC, HID, 1);

    // ---- stage D: GMT pooling per node type (heads=1, all graphs full) ----
    for (int t = 0; t < 2; ++t) {
        const int P = POOLB + 26 * t;
        // PMA1 K/V projections over all nodes
        gemm(xfinb[t], HID, f(P + PL_P1_WK), HID, f(P + PL_P1_BK), Kp, HID, NN, HID, HID, 0);
        gemm(xfinb[t], HID, f(P + PL_P1_WV), HID, f(P + PL_P1_BV), Vp, HID, NN, HID, HID, 0);
        smallvec_lin<<<1, HID, 0, stream>>>(f(P + PL_S1), f(P + PL_P1_WQ), f(P + PL_P1_BQ), qp1);
        pma1_attn<<<BB, 256, 0, stream>>>(qp1, Kp, Vp, h1);
        mab_fco<<<BB, HID, 0, stream>>>(h1, f(P + PL_P1_WO), f(P + PL_P1_BO), h1, HID, 0);
        // SAB (lk=1 -> softmax trivial)
        sab_qv<<<BB, HID, 0, stream>>>(h1, f(P + PL_SB_WQ), f(P + PL_SB_BQ),
                                       f(P + PL_SB_WV), f(P + PL_SB_BV), h2);
        mab_fco<<<BB, HID, 0, stream>>>(h2, f(P + PL_SB_WO), f(P + PL_SB_BO), h2, HID, 0);
        // PMA2 (lk=1)
        smallvec_lin<<<1, HID, 0, stream>>>(f(P + PL_S2), f(P + PL_P2_WQ), f(P + PL_P2_BQ), qp2);
        pma2_qv<<<BB, HID, 0, stream>>>(h2, f(P + PL_P2_WV), f(P + PL_P2_BV), qp2, h1);
        mab_fco<<<BB, HID, 0, stream>>>(h1, f(P + PL_P2_WO), f(P + PL_P2_BO),
                                        hcat, 272, t * HID);
    }

    // ---- stage E: y-MLP into hcat cols 256..271 ----
    ymlp_k<<<1, 256, 0, stream>>>(f(IN_YBASE), f(YMLPB + 0), f(YMLPB + 2),
                                  f(YMLPB + 1), f(YMLPB + 3), hcat);

    // ---- stage F: gmlp (GEMM + BatchNorm over B=16 + GELU) ----
    gmlp_layer<<<1, 256, 0, stream>>>(hcat, 272, f(G_W1), f(G_b1), f(G_g1), f(G_be1), g1b, 256);
    gmlp_layer<<<1, 128, 0, stream>>>(g1b, 256, f(G_W2), f(G_b2), f(G_g2), f(G_be2), g2b, 128);
    gmlp_layer<<<1, 64, 0, stream>>>(g2b, 128, f(G_W3), f(G_b3), f(G_g3), f(G_be3), g3b, 64);
    gmlp_final<<<1, 32, 0, stream>>>(g3b, f(G_W4), f(G_b4), (float*)d_out);
}
